// QLoRABigNet_10565619549059
// MI455X (gfx1250) — matrix operand order repaired and checked by reading, not compiled, not run on hardware
//
#include <hip/hip_runtime.h>
#include <hip/hip_bf16.h>

typedef _Float16 v16h __attribute__((ext_vector_type(16)));
typedef _Float16 h8   __attribute__((ext_vector_type(8)));
typedef _Float16 h4   __attribute__((ext_vector_type(4)));
typedef float    v8f  __attribute__((ext_vector_type(8)));
typedef float    f4   __attribute__((ext_vector_type(4)));
typedef int      i4   __attribute__((ext_vector_type(4)));

#define MDIM 16384
#define NDIM 1024
#define KDIM 1024
#define BM   128
#define BN   128
#define KS   32
#define APAD 40   // halves per LDS row (pad 32 -> 40 for conflict-free b128)
#define BPAD 40

// C[M,1024] = A[M,1024] @ W^T, W dequantized on the fly from int4 groups
// (4 W-rows per group of 4096 elements). Optional addend + relu fused.
// 8 waves: 4 (M) x 2 (N), each wave owns a 32x64 patch = 2x4 WMMA tiles.
template <bool HAS_ADD, bool RELU>
__global__ __launch_bounds__(256)
void gemm_q4(const float* __restrict__ A,
             const int*   __restrict__ wq4,    // [256, 2048] one packed byte per int
             const float* __restrict__ wnorm,  // [256]
             float*       __restrict__ C,
             const float* __restrict__ addend)
{
    __shared__ __align__(16) _Float16 As[BM * APAD];  // 10.0 KB
    __shared__ __align__(16) _Float16 Bs[BN * BPAD];  // 10.0 KB

    const int tid   = threadIdx.x;
    const int lane  = tid & 31;
    const int w     = tid >> 5;
    const int waveM = w >> 1;          // 0..3 -> 32-row band
    const int waveN = w & 1;           // 0..1 -> 64-col band
    const int m0    = blockIdx.y * BM;
    const int n0    = blockIdx.x * BN;

    const int laneh   = lane & 15;     // tile row / column index
    const int lanesel = lane >> 4;     // K-half selector

    v8f acc[2][4];
    #pragma unroll
    for (int tm = 0; tm < 2; ++tm)
        #pragma unroll
        for (int tn = 0; tn < 4; ++tn)
            acc[tm][tn] = (v8f){};

    // A staging: thread -> (row 0..127, 16-wide k segment)
    const int ar   = tid >> 1;
    const int asel = tid & 1;
    const float* Abase = A + (size_t)(m0 + ar) * KDIM + asel * 16;
    // B staging: thread -> (W out-row 0..127, 8-int segment = 16 nibbles)
    const int bo   = tid >> 1;
    const int bi   = tid & 1;
    const int og   = n0 + bo;          // global W output row
    const int grp  = og >> 2;          // int4 group = out_row / 4
    const float nrm   = wnorm[grp];
    const float scale = nrm * (2.0f / 15.0f);
    const float bias  = -nrm;
    const int* qrow = wq4 + grp * 2048 + (og & 3) * 512 + bi * 8;

    // software-pipelined global fetch registers
    f4 apre[4];
    i4 bpre[2];
    {
        const float* Ap = Abase;
        #pragma unroll
        for (int i = 0; i < 4; ++i) apre[i] = *(const f4*)(Ap + i * 4);
        bpre[0] = *(const i4*)(qrow);
        bpre[1] = *(const i4*)(qrow + 4);
    }

    for (int k0 = 0; k0 < KDIM; k0 += KS) {
        __syncthreads();  // previous iteration's LDS reads complete
        // ---- commit prefetched A tile: fp32 -> f16 ----
        {
            _Float16* Ad = As + ar * APAD + asel * 16;
            #pragma unroll
            for (int i = 0; i < 4; ++i) {
                h4 h = { (_Float16)apre[i].x, (_Float16)apre[i].y,
                         (_Float16)apre[i].z, (_Float16)apre[i].w };
                *(h4*)(Ad + i * 4) = h;
            }
        }
        // ---- commit prefetched B tile: int4 dequant -> f16 (Bs[n][k]) ----
        {
            _Float16* Bd = Bs + bo * BPAD + bi * 16;
            #pragma unroll
            for (int half = 0; half < 2; ++half) {
                h8 hb;
                int vals[4] = { bpre[half].x, bpre[half].y, bpre[half].z, bpre[half].w };
                #pragma unroll
                for (int j = 0; j < 4; ++j) {
                    int bv = vals[j];
                    hb[2 * j]     = (_Float16)((float)(bv & 15) * scale + bias);
                    hb[2 * j + 1] = (_Float16)((float)((bv >> 4) & 15) * scale + bias);
                }
                *(h8*)(Bd + half * 8) = hb;
            }
        }
        __syncthreads();

        // ---- prefetch next K-step while WMMAs execute ----
        if (k0 + KS < KDIM) {
            const float* Ap = Abase + (k0 + KS);
            #pragma unroll
            for (int i = 0; i < 4; ++i) apre[i] = *(const f4*)(Ap + i * 4);
            const int* qp = qrow + ((k0 + KS) >> 1);
            bpre[0] = *(const i4*)(qp);
            bpre[1] = *(const i4*)(qp + 4);
        }

        // ---- load fragments & 8 WMMAs ----
        v16h afrag[2], bfrag[4];
        #pragma unroll
        for (int tm = 0; tm < 2; ++tm) {
            // A 16x32 f16: lane m=laneh, K = lanesel*16 + (0..15) contiguous
            const int arow = waveM * 32 + tm * 16 + laneh;
            const _Float16* pa = As + arow * APAD + lanesel * 16;
            h8 alo = *(const h8*)pa;
            h8 ahi = *(const h8*)(pa + 8);
            afrag[tm] = __builtin_shufflevector(alo, ahi,
                         0,1,2,3,4,5,6,7,8,9,10,11,12,13,14,15);
        }
        #pragma unroll
        for (int tn = 0; tn < 4; ++tn) {
            // B 32x16 f16: lane n=laneh, K = lanesel*16 + (0..15) contiguous
            const int nn = waveN * 64 + tn * 16 + laneh;
            const _Float16* pb = Bs + nn * BPAD + lanesel * 16;
            h8 blo = *(const h8*)pb;
            h8 bhi = *(const h8*)(pb + 8);
            bfrag[tn] = __builtin_shufflevector(blo, bhi,
                         0,1,2,3,4,5,6,7,8,9,10,11,12,13,14,15);
        }
        #pragma unroll
        for (int tm = 0; tm < 2; ++tm)
            #pragma unroll
            for (int tn = 0; tn < 4; ++tn)
                acc[tm][tn] = __builtin_amdgcn_wmma_f32_16x16x32_f16(
                    false, afrag[tm], false, bfrag[tn], (short)0, acc[tm][tn],
                    false, false);
    }

    // ---- epilogue: C 16x16 f32 layout: VGPR r -> M = r + 8*lanesel, N = laneh ----
    const int crow0 = m0 + waveM * 32 + lanesel * 8;
    const int ccol0 = n0 + waveN * 64 + laneh;
    #pragma unroll
    for (int tm = 0; tm < 2; ++tm) {
        #pragma unroll
        for (int tn = 0; tn < 4; ++tn) {
            const int col  = ccol0 + tn * 16;
            const int rowb = crow0 + tm * 16;
            #pragma unroll
            for (int r = 0; r < 8; ++r) {
                const size_t off = (size_t)(rowb + r) * NDIM + col;
                float v = acc[tm][tn][r];
                if (HAS_ADD) v += addend[off];
                if (RELU)    v = fmaxf(v, 0.0f);
                C[off] = v;
            }
        }
    }
}

// t[row, 0..32) = h[row,:] @ a^T   (a: [32,1024])
__global__ __launch_bounds__(128)
void lora_t_kernel(const float* __restrict__ h, const float* __restrict__ a,
                   float* __restrict__ t)
{
    __shared__ float hrow[1024];
    __shared__ float part[128];
    const int row = blockIdx.x, tid = threadIdx.x;
    #pragma unroll
    for (int i = 0; i < 8; ++i)
        hrow[tid + i * 128] = h[(size_t)row * 1024 + tid + i * 128];
    __syncthreads();
    const int o = tid >> 2, q = tid & 3;
    const float* ap = a + o * 1024 + q * 256;
    const float* hp = hrow + q * 256;
    float s = 0.0f;
    #pragma unroll 8
    for (int k = 0; k < 256; ++k) s += hp[k] * ap[k];
    part[tid] = s;
    __syncthreads();
    if (q == 0)
        t[(size_t)row * 32 + o] = part[tid] + part[tid + 1] + part[tid + 2] + part[tid + 3];
}

// u[row, col] = t[row,:32] @ b[col,:32]   (b: [1024,32])
__global__ __launch_bounds__(256)
void lora_u_kernel(const float* __restrict__ t, const float* __restrict__ b,
                   float* __restrict__ u)
{
    __shared__ float trow[32];
    const int row = blockIdx.x >> 2;
    const int col = ((blockIdx.x & 3) << 8) + threadIdx.x;
    if (threadIdx.x < 32) trow[threadIdx.x] = t[(size_t)row * 32 + threadIdx.x];
    __syncthreads();
    const float* bp = b + col * 32;
    float s = 0.0f;
    #pragma unroll
    for (int j = 0; j < 32; ++j) s += trow[j] * bp[j];
    u[(size_t)row * 1024 + col] = s;
}

__global__ __launch_bounds__(256)
void ln_kernel(const float* __restrict__ in, const float* __restrict__ gamma,
               const float* __restrict__ beta, float* __restrict__ out)
{
    __shared__ float red[256];
    const int row = blockIdx.x, tid = threadIdx.x;
    float v[4];
    float s = 0.0f;
    #pragma unroll
    for (int i = 0; i < 4; ++i) {
        v[i] = in[(size_t)row * 1024 + tid + i * 256];
        s += v[i];
    }
    red[tid] = s;
    __syncthreads();
    for (int off = 128; off > 0; off >>= 1) {
        if (tid < off) red[tid] += red[tid + off];
        __syncthreads();
    }
    const float mean = red[0] * (1.0f / 1024.0f);
    __syncthreads();
    float s2 = 0.0f;
    #pragma unroll
    for (int i = 0; i < 4; ++i) { float d = v[i] - mean; s2 += d * d; }
    red[tid] = s2;
    __syncthreads();
    for (int off = 128; off > 0; off >>= 1) {
        if (tid < off) red[tid] += red[tid + off];
        __syncthreads();
    }
    const float rstd = rsqrtf(red[0] * (1.0f / 1024.0f) + 1e-5f);
    #pragma unroll
    for (int i = 0; i < 4; ++i) {
        const int c = tid + i * 256;
        out[(size_t)row * 1024 + c] = (v[i] - mean) * rstd * gamma[c] + beta[c];
    }
}

extern "C" void kernel_launch(void* const* d_in, const int* in_sizes, int n_in,
                              void* d_out, int out_size, void* d_ws, size_t ws_size,
                              hipStream_t stream)
{
    (void)in_sizes; (void)n_in; (void)out_size; (void)ws_size;
    const float* x      = (const float*)d_in[0];
    const int*   wq4    = (const int*)  d_in[1];  // [18,256,2048]
    const float* wnorm  = (const float*)d_in[2];  // [18,256]
    const float* lora_a = (const float*)d_in[3];  // [3,32,1024]
    const float* lora_b = (const float*)d_in[4];  // [3,1024,32]
    const float* gam    = (const float*)d_in[5];  // [5,1024]
    const float* bet    = (const float*)d_in[6];  // [5,1024]
    float* out = (float*)d_out;

    const size_t HB = (size_t)MDIM * NDIM;
    float* buf0 = (float*)d_ws;                   // current hidden state
    float* buf1 = buf0 + HB;                      // y0 (also temp t)
    float* buf2 = buf1 + HB;                      // y1 (also LoRA u)

    dim3 ggrid(NDIM / BN, MDIM / BM);             // (8, 128)

    for (int b = 0; b < 6; ++b) {
        const float* in = (b == 0) ? x : buf0;
        const int L = 3 * b;
        if ((b & 1) == 0) {                       // blocks 0,2,4 have QLoRA path
            const int li = b >> 1;
            lora_t_kernel<<<MDIM, 128, 0, stream>>>(in, lora_a + (size_t)li * 32 * 1024, buf1);
            lora_u_kernel<<<MDIM * 4, 256, 0, stream>>>(buf1, lora_b + (size_t)li * 1024 * 32, buf2);
            // y0 = relu(h @ W0^T + u)
            gemm_q4<true, true><<<ggrid, 256, 0, stream>>>(
                in, wq4 + (size_t)L * 256 * 2048, wnorm + (size_t)L * 256, buf1, buf2);
        } else {
            // y0 = relu(h @ W0^T)
            gemm_q4<false, true><<<ggrid, 256, 0, stream>>>(
                in, wq4 + (size_t)L * 256 * 2048, wnorm + (size_t)L * 256, buf1, nullptr);
        }
        // y1 = relu(y0 @ W1^T)
        gemm_q4<false, true><<<ggrid, 256, 0, stream>>>(
            buf1, wq4 + (size_t)(L + 1) * 256 * 2048, wnorm + (size_t)(L + 1) * 256,
            buf2, nullptr);
        // y2 = y1 @ W2^T + h (residual)
        float* dst = (b == 5) ? out : buf1;
        gemm_q4<true, false><<<ggrid, 256, 0, stream>>>(
            buf2, wq4 + (size_t)(L + 2) * 256 * 2048, wnorm + (size_t)(L + 2) * 256,
            dst, in);
        if (b < 5)
            ln_kernel<<<MDIM, 256, 0, stream>>>(buf1, gam + (size_t)b * 1024,
                                                bet + (size_t)b * 1024, buf0);
    }
}
